// GRUModel_1219770712882
// MI455X (gfx1250) — matrix operand order repaired and checked by reading, not compiled, not run on hardware
//
#include <hip/hip_runtime.h>
#include <hip/hip_bf16.h>

// ---------------------------------------------------------------------------
// 2-layer GRU (B=256, T=512, I=128, H=512) + FC(512->1) + sigmoid.
// Persistent pipelined kernel: 16 WGs run layer0, 16 WGs run layer1, coupled
// per-batch-tile by an L2-resident ring buffer + acquire/release flags.
// GEMMs use v_wmma_f32_16x16x32_bf16; hidden state carried in f32 registers.
// ---------------------------------------------------------------------------

#define BB 256
#define TT 512
#define II 128
#define HH 512
#define GG 1536   // 3*H, gate order (r,z,n)

typedef __attribute__((ext_vector_type(16))) __bf16 v16bf;
typedef __attribute__((ext_vector_type(8)))  float  v8f;

union V16U { v16bf v; uint4 u[2]; };
union V8FU { v8f v; float f[8]; };

__device__ __forceinline__ unsigned short f2bf(float x) {
  unsigned int u = __float_as_uint(x);
  unsigned int r = u + 0x7fffu + ((u >> 16) & 1u);   // round-to-nearest-even
  return (unsigned short)(r >> 16);
}

__device__ __forceinline__ float sigm(float x) {
  return 1.0f / (1.0f + __expf(-x));
}

__device__ __forceinline__ v8f wmma_bf16(v16bf a, v16bf b, v8f c) {
  return __builtin_amdgcn_wmma_f32_16x16x32_bf16(false, a, false, b,
                                                 (short)0, c, false, false);
}

// A fragment (16xK tile, bf16, row-major in LDS with row stride `stride`).
// Lane<16: elems 0..7 = K[kb..kb+7], 8..15 = K[kb+16..kb+23], kb=kc*32.
// Lane>=16: same rows, kb=kc*32+8 (covers K 8..15 and 24..31).
__device__ __forceinline__ v16bf load_a(const unsigned short* lds, int stride,
                                        int kc, int lane) {
  int row = lane & 15;
  int kb  = kc * 32 + ((lane >> 4) << 3);
  const unsigned short* p = lds + row * stride + kb;
  V16U r;
  r.u[0] = *(const uint4*)(p);
  r.u[1] = *(const uint4*)(p + 16);
  return r.v;
}

// B fragment (KxN=32x16): B[k][n] = W[n][k] with W row-major [GG][K].
// Lane<16: column n=lane, K[+0..+7] and K[+16..+23]; lane>=16: K[+8..+15], K[+24..+31].
// Two 8-bf16 chunks per lane -> two b128 global loads from L2.
__device__ __forceinline__ v16bf load_b(const unsigned short* w, int K,
                                        int ncol0, int kc, int lane) {
  int n  = ncol0 + (lane & 15);
  int kb = kc * 32 + ((lane >> 4) << 3);
  const unsigned short* p = w + (size_t)n * K + kb;
  V16U r;
  r.u[0] = *(const uint4*)(p);
  r.u[1] = *(const uint4*)(p + 16);
  return r.v;
}

// ---------------------------------------------------------------------------
// One GRU layer for one 16-row batch tile, all T steps (persistent).
// KX = input width (128 for layer0, 512 for layer1). L0 => producer role.
// ---------------------------------------------------------------------------
template <int KX, bool L0>
__device__ void run_layer(int btile, const float* __restrict__ xsrc,
                          const unsigned short* __restrict__ wih,
                          const unsigned short* __restrict__ whh,
                          const float* __restrict__ bih,
                          const float* __restrict__ bhh,
                          float* __restrict__ ring,
                          int* __restrict__ pflag, int* __restrict__ cflag,
                          float* __restrict__ h2last,
                          unsigned short* hstage, unsigned short* xstage) {
  const int tid  = threadIdx.x;
  const int wv   = tid >> 5;       // 16 waves
  const int lane = tid & 31;
  const int cloc = lane & 15;      // column within 16-wide N tile

  // Each wave owns h-column tiles (2*wv) and (2*wv+1) of 32.
  v8f acc_r[2], acc_z[2], acc_in[2], acc_hn[2];
  float hprev[2][8];
  float brz[2], bzz[2], bin_[2], bhn[2];

#pragma unroll
  for (int jt = 0; jt < 2; ++jt) {
    int j = (2 * wv + jt) * 16 + cloc;
    brz[jt]  = bih[j]        + bhh[j];
    bzz[jt]  = bih[512 + j]  + bhh[512 + j];
    bin_[jt] = bih[1024 + j];
    bhn[jt]  = bhh[1024 + j];
#pragma unroll
    for (int v = 0; v < 8; ++v) hprev[jt][v] = 0.0f;
  }

  for (int i = tid; i < 16 * HH; i += 512) hstage[i] = 0;  // h(-1) = 0
  __syncthreads();

  for (int t = 0; t < TT; ++t) {
    // ---- acquire this step's input ----
    if (!L0) {
      if (tid == 0) {
        while (__hip_atomic_load(&pflag[btile], __ATOMIC_ACQUIRE,
                                 __HIP_MEMORY_SCOPE_AGENT) < t + 1)
          __builtin_amdgcn_s_sleep(1);
      }
      __syncthreads();
      const float* slot = ring + (size_t)(t & 3) * BB * HH +
                          (size_t)btile * 16 * HH;
      for (int i = tid; i < 16 * KX; i += 512) xstage[i] = f2bf(slot[i]);
    } else {
      for (int i = tid; i < 16 * KX; i += 512) {
        int row = i / KX, k = i % KX;
        xstage[i] =
            f2bf(xsrc[((size_t)(btile * 16 + row) * TT + t) * II + k]);
      }
    }
    __syncthreads();
    if (!L0 && tid == 0)  // slot consumed into LDS -> free it for producer
      __hip_atomic_store(&cflag[btile], t + 1, __ATOMIC_RELEASE,
                         __HIP_MEMORY_SCOPE_AGENT);

    // ---- GEMMs: gates = [x|h] @ [Wih|Whh]^T ----
    v8f z8 = {0.f, 0.f, 0.f, 0.f, 0.f, 0.f, 0.f, 0.f};
#pragma unroll
    for (int jt = 0; jt < 2; ++jt) {
      acc_r[jt] = z8; acc_z[jt] = z8; acc_in[jt] = z8; acc_hn[jt] = z8;
    }

#pragma unroll 2
    for (int kc = 0; kc < KX / 32; ++kc) {            // input projection
      v16bf a = load_a(xstage, KX, kc, lane);
#pragma unroll
      for (int jt = 0; jt < 2; ++jt) {
        int j0 = (2 * wv + jt) * 16;
        acc_r[jt]  = wmma_bf16(a, load_b(wih, KX, j0,        kc, lane), acc_r[jt]);
        acc_z[jt]  = wmma_bf16(a, load_b(wih, KX, 512 + j0,  kc, lane), acc_z[jt]);
        acc_in[jt] = wmma_bf16(a, load_b(wih, KX, 1024 + j0, kc, lane), acc_in[jt]);
      }
    }
#pragma unroll 2
    for (int kc = 0; kc < HH / 32; ++kc) {            // recurrent projection
      v16bf a = load_a(hstage, HH, kc, lane);
#pragma unroll
      for (int jt = 0; jt < 2; ++jt) {
        int j0 = (2 * wv + jt) * 16;
        acc_r[jt]  = wmma_bf16(a, load_b(whh, HH, j0,        kc, lane), acc_r[jt]);
        acc_z[jt]  = wmma_bf16(a, load_b(whh, HH, 512 + j0,  kc, lane), acc_z[jt]);
        acc_hn[jt] = wmma_bf16(a, load_b(whh, HH, 1024 + j0, kc, lane), acc_hn[jt]);
      }
    }
    __syncthreads();  // all waves done reading hstage for this step

    if (L0) {  // wait until ring slot (t&3) has been consumed
      if (tid == 0 && t >= 4) {
        while (__hip_atomic_load(&cflag[btile], __ATOMIC_ACQUIRE,
                                 __HIP_MEMORY_SCOPE_AGENT) < t - 3)
          __builtin_amdgcn_s_sleep(1);
      }
      __syncthreads();
    }

    // ---- gate math; lanes own exactly the C/D cells they will need next ----
#pragma unroll
    for (int jt = 0; jt < 2; ++jt) {
      int colg = (2 * wv + jt) * 16 + cloc;
      V8FU cr, cz, ci, ch;
      cr.v = acc_r[jt]; cz.v = acc_z[jt]; ci.v = acc_in[jt]; ch.v = acc_hn[jt];
#pragma unroll
      for (int v = 0; v < 8; ++v) {
        int row = v + ((lane >> 4) << 3);
        float r = sigm(cr.f[v] + brz[jt]);
        float z = sigm(cz.f[v] + bzz[jt]);
        float n = tanhf(ci.f[v] + bin_[jt] + r * (ch.f[v] + bhn[jt]));
        float h = (1.0f - z) * n + z * hprev[jt][v];
        hprev[jt][v] = h;
        hstage[row * HH + colg] = f2bf(h);
        if (L0) {
          ring[(size_t)(t & 3) * BB * HH +
               (size_t)(btile * 16 + row) * HH + colg] = h;
        } else if (t == TT - 1) {
          h2last[(size_t)(btile * 16 + row) * HH + colg] = h;
        }
      }
    }
    if (L0) __threadfence();
    __syncthreads();
    if (L0 && tid == 0)
      __hip_atomic_store(&pflag[btile], t + 1, __ATOMIC_RELEASE,
                         __HIP_MEMORY_SCOPE_AGENT);
  }
}

__global__ __launch_bounds__(512) void gru_persistent(
    const float* __restrict__ X, const unsigned short* __restrict__ wih0,
    const unsigned short* __restrict__ whh0, const float* __restrict__ bih0,
    const float* __restrict__ bhh0, const unsigned short* __restrict__ wih1,
    const unsigned short* __restrict__ whh1, const float* __restrict__ bih1,
    const float* __restrict__ bhh1, float* __restrict__ ring,
    int* __restrict__ pflag, int* __restrict__ cflag,
    float* __restrict__ h2last) {
  __shared__ unsigned short hstage[16 * HH];  // 16 KB: h(t-1) in bf16
  __shared__ unsigned short xstage[16 * HH];  // 16 KB: x(t)   in bf16
  int wg = blockIdx.x;
  if (wg < 16) {
    run_layer<II, true>(wg, X, wih0, whh0, bih0, bhh0, ring, pflag, cflag,
                        h2last, hstage, xstage);
  } else {
    run_layer<HH, false>(wg - 16, nullptr, wih1, whh1, bih1, bhh1, ring,
                         pflag, cflag, h2last, hstage, xstage);
  }
}

// ---------------------------------------------------------------------------
__global__ void convert_weights(const float* __restrict__ src,
                                unsigned short* __restrict__ dst, int n) {
  int i = blockIdx.x * blockDim.x + threadIdx.x;
  if (i < n) dst[i] = f2bf(src[i]);
}

__global__ void init_flags(int* pflag, int* cflag) {
  int i = threadIdx.x;
  if (i < 16) { pflag[i] = 0; cflag[i] = 0; }
}

__global__ void fc_kernel(const float* __restrict__ h2,
                          const float* __restrict__ fcw,
                          const float* __restrict__ fcb,
                          float* __restrict__ out) {
  int wv = threadIdx.x >> 5, lane = threadIdx.x & 31;
  int row = blockIdx.x * 8 + wv;
  const float* hp = h2 + (size_t)row * HH;
  float s = 0.0f;
  for (int k = lane; k < HH; k += 32) s += hp[k] * fcw[k];
#pragma unroll
  for (int off = 16; off > 0; off >>= 1) s += __shfl_down(s, off, 32);
  if (lane == 0) out[row] = 1.0f / (1.0f + __expf(-(s + fcb[0])));
}

// ---------------------------------------------------------------------------
extern "C" void kernel_launch(void* const* d_in, const int* in_sizes, int n_in,
                              void* d_out, int out_size, void* d_ws,
                              size_t ws_size, hipStream_t stream) {
  (void)in_sizes; (void)n_in; (void)out_size; (void)ws_size;
  const float* X    = (const float*)d_in[0];
  const float* Wih0 = (const float*)d_in[1];
  const float* Whh0 = (const float*)d_in[2];
  const float* bih0 = (const float*)d_in[3];
  const float* bhh0 = (const float*)d_in[4];
  const float* Wih1 = (const float*)d_in[5];
  const float* Whh1 = (const float*)d_in[6];
  const float* bih1 = (const float*)d_in[7];
  const float* bhh1 = (const float*)d_in[8];
  const float* fcw  = (const float*)d_in[9];
  const float* fcb  = (const float*)d_in[10];
  float* out = (float*)d_out;

  // workspace layout (~7.8 MB total)
  char* p = (char*)d_ws;
  unsigned short* wih0b = (unsigned short*)p; p += (size_t)GG * II * 2;
  unsigned short* whh0b = (unsigned short*)p; p += (size_t)GG * HH * 2;
  unsigned short* wih1b = (unsigned short*)p; p += (size_t)GG * HH * 2;
  unsigned short* whh1b = (unsigned short*)p; p += (size_t)GG * HH * 2;
  float* ring   = (float*)p; p += (size_t)4 * BB * HH * 4;  // 4-deep h1 ring
  float* h2last = (float*)p; p += (size_t)BB * HH * 4;
  int* pflag    = (int*)p;   p += 16 * sizeof(int);
  int* cflag    = (int*)p;   p += 16 * sizeof(int);

  int n0 = GG * II, n1 = GG * HH;
  convert_weights<<<(n0 + 255) / 256, 256, 0, stream>>>(Wih0, wih0b, n0);
  convert_weights<<<(n1 + 255) / 256, 256, 0, stream>>>(Whh0, whh0b, n1);
  convert_weights<<<(n1 + 255) / 256, 256, 0, stream>>>(Wih1, wih1b, n1);
  convert_weights<<<(n1 + 255) / 256, 256, 0, stream>>>(Whh1, whh1b, n1);
  init_flags<<<1, 32, 0, stream>>>(pflag, cflag);

  gru_persistent<<<32, 512, 0, stream>>>(X, wih0b, whh0b, bih0, bhh0, wih1b,
                                         whh1b, bih1, bhh1, ring, pflag,
                                         cflag, h2last);

  fc_kernel<<<32, 256, 0, stream>>>(h2last, fcw, fcb, out);
}
